// RegionProposalNetwork_80195629351017
// MI455X (gfx1250) — matrix operand order repaired and checked by reading, not compiled, not run on hardware
//
#include <hip/hip_runtime.h>
#include <hip/hip_bf16.h>

typedef __attribute__((ext_vector_type(16))) _Float16 v16h;
typedef __attribute__((ext_vector_type(8)))  _Float16 v8h;
typedef __attribute__((ext_vector_type(8)))  float    v8f;

#define BB        2
#define CIN       512
#define HH        50
#define WW2       50
#define SPATIAL   2500      // HH*WW2
#define A0N       35        // 7 ratios * 5 scales
#define NANCH     87500     // SPATIAL * A0N
#define PRE_NMS_K 6000
#define POST_NMS_K 300
#define SORT_N    131072    // next pow2 >= NANCH
#define NSTEP_CONV 144      // 9 taps * 16 cin-blocks
#define NSTEP_HEAD 16       // 16 cin-blocks

// ---------------------------------------------------------------------------
// Anchor generation (matches make_anchors: clipped to [0, img-1])
// ---------------------------------------------------------------------------
__global__ void anchors_kernel(float* __restrict__ anchors_out,
                               const int* __restrict__ imh,
                               const int* __restrict__ imw) {
  int idx = blockIdx.x * blockDim.x + threadIdx.x;
  if (idx >= NANCH) return;
  int s = idx / A0N, k = idx % A0N;
  int y = s / WW2, x = s % WW2;
  int ri = k / 5, sj = k % 5;
  const float ratios[7] = {0.5f, 0.66f, 0.75f, 1.0f, 1.33f, 1.5f, 2.0f};
  const float scales[5] = {2.f, 4.f, 8.f, 16.f, 32.f};
  float r = ratios[ri], sc = scales[sj];
  float h = 16.f * sc * sqrtf(r);
  float w = 16.f * sc * sqrtf(1.f / r);
  float py = 8.f + (float)y * 16.f;
  float px = 8.f + (float)x * 16.f;
  float maxy = (float)(imh[0] - 1);
  float maxx = (float)(imw[0] - 1);
  float y1 = fminf(fmaxf(py - 0.5f * h, 0.f), maxy);
  float x1 = fminf(fmaxf(px - 0.5f * w, 0.f), maxx);
  float y2 = fminf(fmaxf(py + 0.5f * h, 0.f), maxy);
  float x2 = fminf(fmaxf(px + 0.5f * w, 0.f), maxx);
  float* o = anchors_out + (size_t)idx * 4;
  o[0] = y1; o[1] = x1; o[2] = y2; o[3] = x2;
}

// ---------------------------------------------------------------------------
// Fragment-order LDS: [buf][subtile][lane][16 halves].
// A frag (16x32 f16): lane L -> M=L&15; K-octet oct -> g=oct&1, hbase=(oct>>1)*8
// B frag (32x16 f16): lane L -> N=L&15; K-octet oct -> g=oct&1, hbase=(oct>>1)*8
// Per-lane fragment = 32 contiguous bytes (2x ds_load_b128);
// per-thread producer = one v8h ds_store_b128.
// ---------------------------------------------------------------------------
static __device__ __forceinline__ v16h frag_from_lds(const _Float16* p) {
  const v8h* vp = (const v8h*)p;
  v8h lo = vp[0], hi = vp[1];
  return __builtin_shufflevector(lo, hi, 0, 1, 2, 3, 4, 5, 6, 7,
                                 8, 9, 10, 11, 12, 13, 14, 15);
}

static __device__ __forceinline__ void pack_store(_Float16* slot, const float* r) {
  v8h pk;
  #pragma unroll
  for (int q = 0; q < 8; ++q) pk[q] = (_Float16)r[q];
  *(v8h*)slot = pk;
}

// ---------------------------------------------------------------------------
// 3x3 conv 512->512 + bias + ReLU, implicit GEMM via WMMA f16->f32.
// Block 256 thr (8 waves), workgroup tile 64x64, wave tile 16x32 (2 acc).
// Double-buffered LDS: global loads for step+1 issue under step's WMMAs.
// ---------------------------------------------------------------------------
__global__ __launch_bounds__(256) void conv3x3_relu_wmma(
    const float* __restrict__ feat, const float* __restrict__ wgt,
    const float* __restrict__ bias, float* __restrict__ hidden) {
  __shared__ _Float16 aLds[2][4][32][16];   // 8KB
  __shared__ _Float16 bLds[2][4][32][16];   // 8KB
  const int tid  = threadIdx.x;
  const int lane = tid & 31;
  const int wave = tid >> 5;
  const int wm   = wave >> 1;          // 0..3  m-subtile
  const int wn   = wave & 1;           // 0..1  pair of n-subtiles
  const int m0   = blockIdx.x * 64;
  const int n0   = blockIdx.y * 64;
  const int img  = blockIdx.z;

  const int pm   = tid & 63;           // producer row/col 0..63
  const int poct = tid >> 6;           // producer k-octet 0..3

  // per-buffer LDS slots (halves)
  _Float16* aSlot0 = &aLds[0][pm >> 4][(poct & 1) * 16 + (pm & 15)][(poct >> 1) * 8];
  _Float16* bSlot0 = &bLds[0][pm >> 4][(poct & 1) * 16 + (pm & 15)][(poct >> 1) * 8];
  const ptrdiff_t bufStride = 4 * 32 * 16;   // halves between buffers

  const int s  = m0 + pm;
  const int sy = s / WW2, sx = s % WW2;
  const int co = n0 + pm;
  const float* fbase = feat + ((size_t)img * CIN) * (HH * WW2);

  float ra[8], rb[8];

  auto loadStep = [&](int step) {
    const int f  = step >> 4;
    const int cb = (step & 15) * 32 + poct * 8;
    const int ky = f / 3 - 1, kx = f % 3 - 1;
    const int iy = sy + ky, ix = sx + kx;
    const bool ok = (s < SPATIAL) & (iy >= 0) & (iy < HH) & (ix >= 0) & (ix < WW2);
    const int ciy = ok ? iy : 0, cix = ok ? ix : 0;     // clamped safe address
    const float am = ok ? 1.f : 0.f;
    const float* ap = fbase + (size_t)ciy * WW2 + cix;
    #pragma unroll
    for (int q = 0; q < 8; ++q)
      ra[q] = ap[(size_t)(cb + q) * SPATIAL] * am;      // unconditional, clausable
    const float* bp = wgt + (((size_t)co * CIN) * 3 + (ky + 1)) * 3 + (kx + 1);
    #pragma unroll
    for (int q = 0; q < 8; ++q)
      rb[q] = bp[(size_t)(cb + q) * 9];
    if (step + 2 < NSTEP_CONV && ((step + 2) >> 4) == f)
      __builtin_prefetch(ap + (size_t)(cb + 64) * SPATIAL, 0, 1);
  };

  v8f acc0 = {}, acc1 = {};

  loadStep(0);
  pack_store(aSlot0, ra);
  pack_store(bSlot0, rb);
  __syncthreads();

  int buf = 0;
  for (int step = 0; step < NSTEP_CONV; ++step) {
    const bool more = (step + 1 < NSTEP_CONV);
    if (more) loadStep(step + 1);          // global loads in flight under WMMA

    v16h af  = frag_from_lds(&aLds[buf][wm][lane][0]);
    v16h bf0 = frag_from_lds(&bLds[buf][wn * 2 + 0][lane][0]);
    v16h bf1 = frag_from_lds(&bLds[buf][wn * 2 + 1][lane][0]);
    acc0 = __builtin_amdgcn_wmma_f32_16x16x32_f16(false, af, false, bf0,
                                                  (short)0, acc0, false, false);
    acc1 = __builtin_amdgcn_wmma_f32_16x16x32_f16(false, af, false, bf1,
                                                  (short)0, acc1, false, false);
    if (more) {
      const ptrdiff_t off = (buf ^ 1) ? bufStride : 0;
      pack_store(aSlot0 + off, ra);
      pack_store(bSlot0 + off, rb);
    }
    __syncthreads();
    buf ^= 1;
  }

  // C layout: VGPR r -> M = r + 8*(lane>=16), N = lane&15
  const int cmo = (lane >> 4) * 8;
  const int cn0 = n0 + (wn * 2 + 0) * 16 + (lane & 15);
  const int cn1 = n0 + (wn * 2 + 1) * 16 + (lane & 15);
  #pragma unroll
  for (int r = 0; r < 8; ++r) {
    int so = m0 + wm * 16 + cmo + r;
    if (so < SPATIAL) {
      hidden[((size_t)img * CIN + cn0) * SPATIAL + so] = fmaxf(acc0[r] + bias[cn0], 0.f);
      hidden[((size_t)img * CIN + cn1) * SPATIAL + so] = fmaxf(acc1[r] + bias[cn1], 0.f);
    }
  }
}

// ---------------------------------------------------------------------------
// Fused 1x1 loc(140)+score(70) convs: GEMM M=2500/img, N=210 (pad 256), K=512.
// ---------------------------------------------------------------------------
__global__ __launch_bounds__(256) void head1x1_wmma(
    const float* __restrict__ hidden,
    const float* __restrict__ loc_w, const float* __restrict__ loc_b,
    const float* __restrict__ score_w, const float* __restrict__ score_b,
    float* __restrict__ out_locs, float* __restrict__ out_scores) {
  __shared__ _Float16 aLds[2][4][32][16];
  __shared__ _Float16 bLds[2][4][32][16];
  const int tid  = threadIdx.x;
  const int lane = tid & 31;
  const int wave = tid >> 5;
  const int wm   = wave >> 1;
  const int wn   = wave & 1;
  const int m0   = blockIdx.x * 64;
  const int n0   = blockIdx.y * 64;
  const int img  = blockIdx.z;

  const int pm   = tid & 63;
  const int poct = tid >> 6;

  _Float16* aSlot0 = &aLds[0][pm >> 4][(poct & 1) * 16 + (pm & 15)][(poct >> 1) * 8];
  _Float16* bSlot0 = &bLds[0][pm >> 4][(poct & 1) * 16 + (pm & 15)][(poct >> 1) * 8];
  const ptrdiff_t bufStride = 4 * 32 * 16;

  const int s = m0 + pm;
  const bool aok = (s < SPATIAL);
  const float am = aok ? 1.f : 0.f;
  const float* ap = hidden + ((size_t)img * CIN) * SPATIAL + (aok ? s : 0);

  // weight pointer select (uniform per thread), clamped, masked
  const int co = n0 + pm;
  float bm = 1.f;
  const float* wp;
  if (co < 140) {
    wp = loc_w + (size_t)co * CIN;
  } else {
    int scid = co - 140;
    if (scid >= 70) { scid = 0; bm = 0.f; }
    wp = score_w + (size_t)scid * CIN;
  }

  float ra[8], rb[8];
  auto loadStep = [&](int step) {
    const int cb = step * 32 + poct * 8;
    #pragma unroll
    for (int q = 0; q < 8; ++q)
      ra[q] = ap[(size_t)(cb + q) * SPATIAL] * am;
    #pragma unroll
    for (int q = 0; q < 8; ++q)
      rb[q] = wp[cb + q] * bm;
    if (step + 2 < NSTEP_HEAD)
      __builtin_prefetch(ap + (size_t)(cb + 64) * SPATIAL, 0, 1);
  };

  v8f acc0 = {}, acc1 = {};

  loadStep(0);
  pack_store(aSlot0, ra);
  pack_store(bSlot0, rb);
  __syncthreads();

  int buf = 0;
  for (int step = 0; step < NSTEP_HEAD; ++step) {
    const bool more = (step + 1 < NSTEP_HEAD);
    if (more) loadStep(step + 1);

    v16h af  = frag_from_lds(&aLds[buf][wm][lane][0]);
    v16h bf0 = frag_from_lds(&bLds[buf][wn * 2 + 0][lane][0]);
    v16h bf1 = frag_from_lds(&bLds[buf][wn * 2 + 1][lane][0]);
    acc0 = __builtin_amdgcn_wmma_f32_16x16x32_f16(false, af, false, bf0,
                                                  (short)0, acc0, false, false);
    acc1 = __builtin_amdgcn_wmma_f32_16x16x32_f16(false, af, false, bf1,
                                                  (short)0, acc1, false, false);
    if (more) {
      const ptrdiff_t off = (buf ^ 1) ? bufStride : 0;
      pack_store(aSlot0 + off, ra);
      pack_store(bSlot0 + off, rb);
    }
    __syncthreads();
    buf ^= 1;
  }

  const int cmo = (lane >> 4) * 8;
  #pragma unroll
  for (int half = 0; half < 2; ++half) {
    const int cn = n0 + (wn * 2 + half) * 16 + (lane & 15);
    const v8f& acc = half ? acc1 : acc0;
    #pragma unroll
    for (int r = 0; r < 8; ++r) {
      int so = m0 + wm * 16 + cmo + r;
      if (so < SPATIAL && cn < 210) {
        if (cn < 140) {
          out_locs[(size_t)img * (NANCH * 4) + (size_t)so * 140 + cn] =
              acc[r] + loc_b[cn];
        } else {
          out_scores[(size_t)img * (NANCH * 2) + (size_t)so * 70 + (cn - 140)] =
              acc[r] + score_b[cn - 140];
        }
      }
    }
  }
}

// ---------------------------------------------------------------------------
// Per-anchor: softmax fg, loc2bbox decode, clip, min-size filter.
// ---------------------------------------------------------------------------
__global__ void decode_kernel(const float* __restrict__ locs,
                              const float* __restrict__ scores,
                              const float* __restrict__ anchors,
                              float* __restrict__ boxes,
                              float* __restrict__ scAdj,
                              const int* __restrict__ imh,
                              const int* __restrict__ imw) {
  int t = blockIdx.x * blockDim.x + threadIdx.x;
  if (t >= BB * NANCH) return;
  int img = t / NANCH, i = t % NANCH;
  const float* a = anchors + (size_t)i * 4;
  float ah = a[2] - a[0], aw = a[3] - a[1];
  float acy = a[0] + 0.5f * ah, acx = a[1] + 0.5f * aw;
  const float* l = locs + (size_t)img * NANCH * 4 + (size_t)i * 4;
  float cy = l[0] * ah + acy;
  float cx = l[1] * aw + acx;
  float hh = __expf(l[2]) * ah;
  float ww = __expf(l[3]) * aw;
  float fh = (float)imh[0], fw = (float)imw[0];
  float y1 = fminf(fmaxf(cy - 0.5f * hh, 0.f), fh);
  float x1 = fminf(fmaxf(cx - 0.5f * ww, 0.f), fw);
  float y2 = fminf(fmaxf(cy + 0.5f * hh, 0.f), fh);
  float x2 = fminf(fmaxf(cx + 0.5f * ww, 0.f), fw);
  float* bo = boxes + (size_t)t * 4;
  bo[0] = y1; bo[1] = x1; bo[2] = y2; bo[3] = x2;
  bool valid = (y2 - y1 >= 16.f) && (x2 - x1 >= 16.f);
  float s0 = scores[(size_t)img * NANCH * 2 + (size_t)i * 2 + 0];
  float s1 = scores[(size_t)img * NANCH * 2 + (size_t)i * 2 + 1];
  float fg = 1.f / (1.f + __expf(s0 - s1));   // softmax fg channel
  scAdj[t] = valid ? fg : -__builtin_inff();
}

// ---------------------------------------------------------------------------
// Bitonic sort (descending by key) over padded 131072 keys per image.
// ---------------------------------------------------------------------------
__global__ void sort_init_kernel(const float* __restrict__ scAdj,
                                 float* __restrict__ keys, int* __restrict__ vals) {
  int t = blockIdx.x * blockDim.x + threadIdx.x;
  if (t >= BB * SORT_N) return;
  int img = t >> 17, i = t & (SORT_N - 1);
  keys[t] = (i < NANCH) ? scAdj[(size_t)img * NANCH + i] : -__builtin_inff();
  vals[t] = i;
}

__global__ void bitonic_pass(float* __restrict__ keys, int* __restrict__ vals,
                             int k, int j) {
  int t = blockIdx.x * blockDim.x + threadIdx.x;
  if (t >= BB * SORT_N) return;
  int img = t >> 17;
  int i = t & (SORT_N - 1);
  int l = i ^ j;
  if (l <= i) return;
  float* K = keys + (size_t)img * SORT_N;
  int*   V = vals + (size_t)img * SORT_N;
  bool desc = ((i & k) == 0);
  float ki = K[i], kl = K[l];
  bool doswap = desc ? (ki < kl) : (ki > kl);
  if (doswap) {
    K[i] = kl; K[l] = ki;
    int vi = V[i]; V[i] = V[l]; V[l] = vi;
  }
}

__global__ void gather_top_kernel(const float* __restrict__ keys,
                                  const int* __restrict__ vals,
                                  const float* __restrict__ boxes,
                                  float* __restrict__ topBoxes,
                                  int* __restrict__ keepInit) {
  int t = blockIdx.x * blockDim.x + threadIdx.x;
  if (t >= BB * PRE_NMS_K) return;
  int img = t / PRE_NMS_K, r = t % PRE_NMS_K;
  float key = keys[(size_t)img * SORT_N + r];
  int src = vals[(size_t)img * SORT_N + r];
  if (src >= NANCH) src = 0;
  bool valid = key > -1e30f;
  const float* sb = boxes + ((size_t)img * NANCH + src) * 4;
  float* db = topBoxes + (size_t)t * 4;
  db[0] = sb[0]; db[1] = sb[1]; db[2] = sb[2]; db[3] = sb[3];
  keepInit[t] = valid ? 1 : 0;
}

// ---------------------------------------------------------------------------
// NMS (sequential, matches reference fori_loop) + top-300 finalize.
// ---------------------------------------------------------------------------
__global__ __launch_bounds__(1024) void nms_finalize_kernel(
    const float* __restrict__ topBoxes, const int* __restrict__ keepInit,
    float* __restrict__ rois, float* __restrict__ roi_idx,
    float* __restrict__ roi_valid) {
  __shared__ unsigned char keep[PRE_NMS_K];
  __shared__ int sel[POST_NMS_K];
  __shared__ int nsel;
  const int img = blockIdx.x;
  const int tid = threadIdx.x;
  const float* B = topBoxes + (size_t)img * PRE_NMS_K * 4;

  for (int i = tid; i < PRE_NMS_K; i += blockDim.x)
    keep[i] = (unsigned char)keepInit[img * PRE_NMS_K + i];
  __syncthreads();

  for (int i = 0; i < PRE_NMS_K; ++i) {
    if (keep[i]) {   // uniform: LDS read after barrier of previous iteration
      float iy1 = B[i * 4 + 0], ix1 = B[i * 4 + 1];
      float iy2 = B[i * 4 + 2], ix2 = B[i * 4 + 3];
      float ia = (iy2 - iy1) * (ix2 - ix1);
      for (int j = i + 1 + tid; j < PRE_NMS_K; j += blockDim.x) {
        if (!keep[j]) continue;
        float jy1 = B[j * 4 + 0], jx1 = B[j * 4 + 1];
        float jy2 = B[j * 4 + 2], jx2 = B[j * 4 + 3];
        float ty = fmaxf(iy1, jy1), tx = fmaxf(ix1, jx1);
        float by = fminf(iy2, jy2), bx = fminf(ix2, jx2);
        float ih = fmaxf(by - ty, 0.f), iw = fmaxf(bx - tx, 0.f);
        float inter = ih * iw;
        float ja = (jy2 - jy1) * (jx2 - jx1);
        float iou = inter / (ia + ja - inter + 1e-9f);
        if (iou > 0.7f) keep[j] = 0;
      }
    }
    __syncthreads();
  }

  if (tid == 0) {
    int c = 0;
    for (int i = 0; i < PRE_NMS_K && c < POST_NMS_K; ++i)
      if (keep[i]) sel[c++] = i;
    nsel = c;
  }
  __syncthreads();

  for (int p = tid; p < POST_NMS_K; p += blockDim.x) {
    bool v = p < nsel;
    int sidx = v ? sel[p] : 0;
    float* ro = rois + ((size_t)img * POST_NMS_K + p) * 4;
    ro[0] = v ? B[sidx * 4 + 0] : 0.f;
    ro[1] = v ? B[sidx * 4 + 1] : 0.f;
    ro[2] = v ? B[sidx * 4 + 2] : 0.f;
    ro[3] = v ? B[sidx * 4 + 3] : 0.f;
    roi_idx[img * POST_NMS_K + p]   = (float)img;
    roi_valid[img * POST_NMS_K + p] = v ? 1.f : 0.f;
  }
}

// ---------------------------------------------------------------------------
extern "C" void kernel_launch(void* const* d_in, const int* in_sizes, int n_in,
                              void* d_out, int out_size, void* d_ws, size_t ws_size,
                              hipStream_t stream) {
  const float* feat    = (const float*)d_in[0];
  const float* conv_w  = (const float*)d_in[1];
  const float* conv_b  = (const float*)d_in[2];
  const float* score_w = (const float*)d_in[3];
  const float* score_b = (const float*)d_in[4];
  const float* loc_w   = (const float*)d_in[5];
  const float* loc_b   = (const float*)d_in[6];
  const int*   imh     = (const int*)d_in[7];
  const int*   imw     = (const int*)d_in[8];

  // Output layout (floats, flat, return order)
  float* out         = (float*)d_out;
  float* out_locs    = out;                 // B*87500*4 = 700000
  float* out_scores  = out + 700000;        // B*87500*2 = 350000
  float* out_rois    = out + 1050000;       // B*300*4   = 2400
  float* out_ridx    = out + 1052400;       // B*300     = 600
  float* out_rvalid  = out + 1053000;       // B*300     = 600
  float* out_anchors = out + 1053600;       // 87500*4   = 350000

  // Workspace layout
  float* ws       = (float*)d_ws;
  float* hidden   = ws;                         // B*512*2500 = 2,560,000
  float* boxes    = hidden + 2560000;           // B*87500*4  =   700,000
  float* scAdj    = boxes + 700000;             // B*87500    =   175,000
  float* keys     = scAdj + 175000;             // B*131072   =   262,144
  int*   vals     = (int*)(keys + 262144);      // B*131072   =   262,144
  float* topBoxes = (float*)(vals + 262144);    // B*6000*4   =    48,000
  int*   keepInit = (int*)(topBoxes + 48000);   // B*6000     =    12,000

  anchors_kernel<<<(NANCH + 255) / 256, 256, 0, stream>>>(out_anchors, imh, imw);

  conv3x3_relu_wmma<<<dim3(40, 8, BB), 256, 0, stream>>>(feat, conv_w, conv_b, hidden);

  head1x1_wmma<<<dim3(40, 4, BB), 256, 0, stream>>>(hidden, loc_w, loc_b, score_w,
                                                    score_b, out_locs, out_scores);

  decode_kernel<<<(BB * NANCH + 255) / 256, 256, 0, stream>>>(
      out_locs, out_scores, out_anchors, boxes, scAdj, imh, imw);

  sort_init_kernel<<<(BB * SORT_N) / 256, 256, 0, stream>>>(scAdj, keys, vals);
  for (int k = 2; k <= SORT_N; k <<= 1)
    for (int j = k >> 1; j >= 1; j >>= 1)
      bitonic_pass<<<(BB * SORT_N) / 256, 256, 0, stream>>>(keys, vals, k, j);

  gather_top_kernel<<<(BB * PRE_NMS_K + 255) / 256, 256, 0, stream>>>(
      keys, vals, boxes, topBoxes, keepInit);

  nms_finalize_kernel<<<BB, 1024, 0, stream>>>(topBoxes, keepInit, out_rois,
                                               out_ridx, out_rvalid);
}
